// MoE_77841987273023
// MI455X (gfx1250) — matrix operand order repaired and checked by reading, not compiled, not run on hardware
//
#include <hip/hip_runtime.h>
#include <hip/hip_bf16.h>
#include <math.h>

// Problem constants (from reference): B=2,S=2048 -> N=4096 tokens
#define NTOK 4096
#define HDIM 1024
#define NEXP 16
#define IDIM 512
#define ISH  2048
#define TOPK 4

typedef __attribute__((ext_vector_type(16))) __bf16 v16bf;
typedef __attribute__((ext_vector_type(8)))  float  v8f;
typedef __attribute__((ext_vector_type(4)))  int    i4;

#define MBLK 128
#define KSTEP 64
#define LDK (KSTEP + 8)   // LDS row padding (16B) against bank conflicts

union FragU  { v16bf v; uint4 q[2]; };
union Pack16 { __bf16 h[16]; uint4 q[2]; };
union Pack8  { __bf16 h[8];  uint4 q; };

// ---------------- CDNA5 async global->LDS (ASYNCcnt-tracked DMA) -----------
typedef __attribute__((address_space(1))) i4* g4ptr_t;   // global int32x4*
typedef __attribute__((address_space(3))) i4* l4ptr_t;   // LDS int32x4*

__device__ __forceinline__ void async_copy16(void* lds, const void* gsrc) {
#if __has_builtin(__builtin_amdgcn_global_load_async_to_lds_b128)
  __builtin_amdgcn_global_load_async_to_lds_b128((g4ptr_t)gsrc, (l4ptr_t)lds, 0, 0);
#else
  unsigned loff = (unsigned)(unsigned long long)(l4ptr_t)lds;
  asm volatile("global_load_async_to_lds_b128 %0, %1, off"
               :: "v"(loff), "v"(gsrc) : "memory");
#endif
}

__device__ __forceinline__ void wait_async0() {
#if __has_builtin(__builtin_amdgcn_s_wait_asynccnt)
  __builtin_amdgcn_s_wait_asynccnt(0);
#else
  asm volatile("s_wait_asynccnt 0" ::: "memory");
#endif
}

// ---------------- WMMA fragment loaders (per CDNA5 ISA lane layouts) -------
// A 16x32 bf16: lanes 0-15 row m=lane, K {0..7,16..23}; lanes 16-31 K {8..15,24..31}
__device__ __forceinline__ v16bf lds_a_frag(const __bf16* lds, int mrow, int kk) {
  int lane = threadIdx.x & 31;
  int lo = (lane < 16) ? 0 : 8;
  const __bf16* base = lds + (mrow + (lane & 15)) * LDK + kk + lo;
  FragU f;
  f.q[0] = *(const uint4*)(base);
  f.q[1] = *(const uint4*)(base + 16);
  return f.v;
}
// B 32x16 bf16: lanes 0-15 col n=lane, K {0..7,16..23}; lanes 16-31 K {8..15,24..31}
__device__ __forceinline__ v16bf lds_b_frag(const __bf16* lds, int ncol, int kk) {
  int lane = threadIdx.x & 31;
  int kh = (lane < 16) ? 0 : 8;
  const __bf16* base = lds + (ncol + (lane & 15)) * LDK + kk + kh;
  FragU f;
  f.q[0] = *(const uint4*)(base);
  f.q[1] = *(const uint4*)(base + 16);
  return f.v;
}

__device__ __forceinline__ v8f wmma_bf16(v16bf a, v16bf b, v8f c) {
  return __builtin_amdgcn_wmma_f32_16x16x32_bf16(false, a, false, b, (short)0, c,
                                                 false, false);
}

__device__ __forceinline__ v8f vzero8() {
  v8f z = {0.f, 0.f, 0.f, 0.f, 0.f, 0.f, 0.f, 0.f};
  return z;
}

__device__ __forceinline__ float silu_f(float x) {
  return x / (1.0f + __expf(-x));
}

// Stage a 128x64 bf16 tile (global row-major, stride in elems) into LDS
// using the CDNA5 async global->LDS path (16B per lane per op).
__device__ __forceinline__ void stage_a_async(__bf16* dst, const __bf16* src,
                                              size_t stride) {
  int t = threadIdx.x;
  int r = t >> 1, c = (t & 1) * 32;
  const __bf16* s = src + (size_t)r * stride + c;
  __bf16* d = dst + r * LDK + c;
#pragma unroll
  for (int i = 0; i < 4; i++) async_copy16(d + i * 8, s + i * 8);
}

// Gathered variant: per-thread source row pointer precomputed by caller.
__device__ __forceinline__ void stage_a_async_row(__bf16* dstRow,
                                                  const __bf16* srcRow) {
#pragma unroll
  for (int i = 0; i < 4; i++) async_copy16(dstRow + i * 8, srcRow + i * 8);
}

// Stage a ROWSx64 f32 weight tile into LDS as bf16 (on-the-fly conversion),
// prefetching the next K tile of the weight stream into cache.
template <int ROWS>
__device__ __forceinline__ void load_wtile_f32(__bf16* dst, const float* src,
                                               size_t stride, bool pf) {
  int t = threadIdx.x;
  constexpr int TPR = 256 / ROWS;       // threads per row
  constexpr int CPT = KSTEP / TPR;      // f32 elements per thread
  int r = t / TPR, c = (t % TPR) * CPT;
  const float* sp = src + (size_t)r * stride + c;
  __bf16* dp = dst + r * LDK + c;
  if (pf) __builtin_prefetch(sp + KSTEP, 0, 3);   // global_prefetch_b8
#pragma unroll
  for (int j = 0; j < CPT; j += 16) {
    const float4* s = (const float4*)(sp + j);
    Pack16 pk;
#pragma unroll
    for (int i = 0; i < 4; i++) {
      float4 f = s[i];
      pk.h[i * 4 + 0] = (__bf16)f.x; pk.h[i * 4 + 1] = (__bf16)f.y;
      pk.h[i * 4 + 2] = (__bf16)f.z; pk.h[i * 4 + 3] = (__bf16)f.w;
    }
    uint4* d = (uint4*)(dp + j);
    d[0] = pk.q[0]; d[1] = pk.q[1];
  }
}

// ---------------- 1. hidden f32 -> bf16 ------------------------------------
__global__ __launch_bounds__(256) void k_cvt(const float* __restrict__ h,
                                             __bf16* __restrict__ hb) {
  size_t i = ((size_t)blockIdx.x * 256 + threadIdx.x) * 8;
  float4 a = *(const float4*)(h + i);
  float4 b = *(const float4*)(h + i + 4);
  Pack8 p;
  p.h[0] = (__bf16)a.x; p.h[1] = (__bf16)a.y; p.h[2] = (__bf16)a.z; p.h[3] = (__bf16)a.w;
  p.h[4] = (__bf16)b.x; p.h[5] = (__bf16)b.y; p.h[6] = (__bf16)b.z; p.h[7] = (__bf16)b.w;
  *(uint4*)(hb + i) = p.q;
}

// ---------------- 2. router: logits -> softmax -> top4 -> renorm -----------
__global__ __launch_bounds__(256) void k_router(const float* __restrict__ h,
                                                const float* __restrict__ rw,
                                                const float* __restrict__ segw,
                                                int* __restrict__ topi,
                                                float* __restrict__ topw,
                                                float* __restrict__ gatev) {
  int n = blockIdx.x * 256 + threadIdx.x;
  const float* hr = h + (size_t)n * HDIM;
  float acc[NEXP];
#pragma unroll
  for (int e = 0; e < NEXP; e++) acc[e] = 0.f;
  float accg = 0.f;
  for (int k = 0; k < HDIM; k += 4) {
    float4 hv = *(const float4*)(hr + k);
#pragma unroll
    for (int e = 0; e < NEXP; e++) {
      float4 wv = *(const float4*)(rw + (size_t)e * HDIM + k);
      acc[e] += hv.x * wv.x + hv.y * wv.y + hv.z * wv.z + hv.w * wv.w;
    }
    float4 gw = *(const float4*)(segw + k);
    accg += hv.x * gw.x + hv.y * gw.y + hv.z * gw.z + hv.w * gw.w;
  }
  float mx = acc[0];
#pragma unroll
  for (int e = 1; e < NEXP; e++) mx = fmaxf(mx, acc[e]);
  unsigned usedMask = 0u;
  int ids[TOPK]; float ex[TOPK];
  float wsum = 0.f;
#pragma unroll
  for (int kk = 0; kk < TOPK; kk++) {
    int bi = 0; float bv = -3.4e38f;
#pragma unroll
    for (int e = 0; e < NEXP; e++) {
      bool ok = !((usedMask >> e) & 1u) && (acc[e] > bv);
      bv = ok ? acc[e] : bv;
      bi = ok ? e : bi;
    }
    usedMask |= 1u << bi;
    ids[kk] = bi;
    ex[kk] = __expf(acc[bi] - mx);
    wsum += ex[kk];
  }
  float inv = 1.0f / wsum;
#pragma unroll
  for (int kk = 0; kk < TOPK; kk++) {
    topi[n * TOPK + kk] = ids[kk];
    topw[n * TOPK + kk] = ex[kk] * inv;
  }
  gatev[n] = 1.0f / (1.0f + __expf(-accg));
}

// ---------------- 3. deterministic token compaction (single block) ---------
__global__ __launch_bounds__(256) void k_compact(const int* __restrict__ topi,
                                                 const float* __restrict__ topw,
                                                 int* __restrict__ counts,
                                                 int* __restrict__ offs,
                                                 int* __restrict__ slot_tok,
                                                 float* __restrict__ slot_w,
                                                 int* __restrict__ slot_of) {
  __shared__ int cnt[NEXP][256];
  __shared__ int tot[NEXP];
  __shared__ int offsh[NEXP + 1];
  int t = threadIdx.x;
  constexpr int CHUNK = NTOK / 256;
#pragma unroll
  for (int e = 0; e < NEXP; e++) cnt[e][t] = 0;
  __syncthreads();
  int n0 = t * CHUNK;
  for (int i = 0; i < CHUNK; i++)
    for (int k = 0; k < TOPK; k++)
      cnt[topi[(n0 + i) * TOPK + k]][t]++;
  __syncthreads();
  if (t < NEXP) {                       // exclusive scan across threads
    int run = 0;
    for (int j = 0; j < 256; j++) { int v = cnt[t][j]; cnt[t][j] = run; run += v; }
    tot[t] = run;
  }
  __syncthreads();
  if (t == 0) {
    int o = 0;
    for (int e = 0; e < NEXP; e++) { offsh[e] = o; o += tot[e]; }
    offsh[NEXP] = o;
    for (int e = 0; e < NEXP; e++) { counts[e] = tot[e]; offs[e] = offsh[e]; }
    offs[NEXP] = o;
  }
  __syncthreads();
  for (int i = 0; i < CHUNK; i++)
    for (int k = 0; k < TOPK; k++) {
      int idx = (n0 + i) * TOPK + k;
      int e = topi[idx];
      int s = offsh[e] + cnt[e][t]++;   // cnt row reused as cursor
      slot_tok[s] = n0 + i;
      slot_w[s] = topw[idx];
      slot_of[idx] = s;
    }
}

// ---------------- 4. shared expert: silu(h@Wg.T) * (h@Wu.T) ----------------
__global__ __launch_bounds__(256) void k_shared_gateup(
    const __bf16* __restrict__ hb, const float* __restrict__ Wg,
    const float* __restrict__ Wu, __bf16* __restrict__ act) {
  __shared__ __align__(16) __bf16 As[MBLK * LDK];
  __shared__ __align__(16) __bf16 Bg[64 * LDK];
  __shared__ __align__(16) __bf16 Bu[64 * LDK];
  int wave = threadIdx.x >> 5;
  int m0 = blockIdx.y * MBLK;
  int n0 = blockIdx.x * 64;
  v8f cg[4], cu[4];
#pragma unroll
  for (int j = 0; j < 4; j++) { cg[j] = vzero8(); cu[j] = vzero8(); }
  for (int k0 = 0; k0 < HDIM; k0 += KSTEP) {
    stage_a_async(As, hb + (size_t)m0 * HDIM + k0, HDIM);   // DMA, ASYNCcnt
    load_wtile_f32<64>(Bg, Wg + (size_t)n0 * HDIM + k0, HDIM, k0 + KSTEP < HDIM);
    load_wtile_f32<64>(Bu, Wu + (size_t)n0 * HDIM + k0, HDIM, k0 + KSTEP < HDIM);
    wait_async0();
    __syncthreads();
#pragma unroll
    for (int kk = 0; kk < KSTEP; kk += 32) {
      v16bf a = lds_a_frag(As, wave * 16, kk);
#pragma unroll
      for (int j = 0; j < 4; j++) {
        cg[j] = wmma_bf16(a, lds_b_frag(Bg, j * 16, kk), cg[j]);
        cu[j] = wmma_bf16(a, lds_b_frag(Bu, j * 16, kk), cu[j]);
      }
    }
    __syncthreads();
  }
  int lane = threadIdx.x & 31;
  int col = n0 + (lane & 15);
  int rb = m0 + wave * 16 + ((lane & 16) ? 8 : 0);
#pragma unroll
  for (int j = 0; j < 4; j++)
#pragma unroll
    for (int r = 0; r < 8; r++) {
      float a = silu_f(cg[j][r]) * cu[j][r];
      act[(size_t)(rb + r) * ISH + col + j * 16] = (__bf16)a;
    }
}

// ---------------- 5. shared expert down proj (+ sigmoid gate) -> d_out -----
__global__ __launch_bounds__(256) void k_shared_down(
    const __bf16* __restrict__ act, const float* __restrict__ Wd,
    const float* __restrict__ gatev, float* __restrict__ out) {
  __shared__ __align__(16) __bf16 As[MBLK * LDK];
  __shared__ __align__(16) __bf16 Bs[MBLK * LDK];
  int wave = threadIdx.x >> 5;
  int m0 = blockIdx.y * MBLK;
  int n0 = blockIdx.x * 128;
  v8f c[8];
#pragma unroll
  for (int j = 0; j < 8; j++) c[j] = vzero8();
  for (int k0 = 0; k0 < ISH; k0 += KSTEP) {
    stage_a_async(As, act + (size_t)m0 * ISH + k0, ISH);
    load_wtile_f32<128>(Bs, Wd + (size_t)n0 * ISH + k0, ISH, k0 + KSTEP < ISH);
    wait_async0();
    __syncthreads();
#pragma unroll
    for (int kk = 0; kk < KSTEP; kk += 32) {
      v16bf a = lds_a_frag(As, wave * 16, kk);
#pragma unroll
      for (int j = 0; j < 8; j++)
        c[j] = wmma_bf16(a, lds_b_frag(Bs, j * 16, kk), c[j]);
    }
    __syncthreads();
  }
  int lane = threadIdx.x & 31;
  int col = n0 + (lane & 15);
  int rb = m0 + wave * 16 + ((lane & 16) ? 8 : 0);
#pragma unroll
  for (int r = 0; r < 8; r++) {
    float g = gatev[rb + r];
#pragma unroll
    for (int j = 0; j < 8; j++)
      out[(size_t)(rb + r) * HDIM + col + j * 16] = g * c[j][r];
  }
}

// ---------------- 6. routed experts: gathered gate/up + SiLU ---------------
__global__ __launch_bounds__(256) void k_expert_gateup(
    const __bf16* __restrict__ hb, const float* __restrict__ gup,
    const int* __restrict__ counts, const int* __restrict__ offs,
    const int* __restrict__ slot_tok, __bf16* __restrict__ acte) {
  int e = blockIdx.z;
  int count = counts[e];
  int m0 = blockIdx.y * MBLK;
  if (m0 >= count) return;
  int base = offs[e];
  const float* W = gup + (size_t)e * (2 * IDIM) * HDIM;
  int n0 = blockIdx.x * 64;
  __shared__ __align__(16) __bf16 As[MBLK * LDK];
  __shared__ __align__(16) __bf16 Bg[64 * LDK];
  __shared__ __align__(16) __bf16 Bu[64 * LDK];
  int wave = threadIdx.x >> 5;
  // gathered A row for this thread (fixed across K loop)
  int lr = threadIdx.x >> 1, lc = (threadIdx.x & 1) * 32;
  int lidx = m0 + lr;
  int tok = (lidx < count) ? slot_tok[base + lidx] : 0;
  const __bf16* arow = hb + (size_t)tok * HDIM + lc;
  __bf16* drow = As + lr * LDK + lc;
  v8f cg[4], cu[4];
#pragma unroll
  for (int j = 0; j < 4; j++) { cg[j] = vzero8(); cu[j] = vzero8(); }
  for (int k0 = 0; k0 < HDIM; k0 += KSTEP) {
    stage_a_async_row(drow, arow + k0);
    load_wtile_f32<64>(Bg, W + (size_t)n0 * HDIM + k0, HDIM, k0 + KSTEP < HDIM);
    load_wtile_f32<64>(Bu, W + (size_t)(IDIM + n0) * HDIM + k0, HDIM,
                       k0 + KSTEP < HDIM);
    wait_async0();
    __syncthreads();
#pragma unroll
    for (int kk = 0; kk < KSTEP; kk += 32) {
      v16bf a = lds_a_frag(As, wave * 16, kk);
#pragma unroll
      for (int j = 0; j < 4; j++) {
        cg[j] = wmma_bf16(a, lds_b_frag(Bg, j * 16, kk), cg[j]);
        cu[j] = wmma_bf16(a, lds_b_frag(Bu, j * 16, kk), cu[j]);
      }
    }
    __syncthreads();
  }
  int lane = threadIdx.x & 31;
  int col = n0 + (lane & 15);
  int rb = wave * 16 + ((lane & 16) ? 8 : 0) + m0;
#pragma unroll
  for (int r = 0; r < 8; r++) {
    int row = rb + r;
    if (row < count) {
#pragma unroll
      for (int j = 0; j < 4; j++) {
        float a = silu_f(cg[j][r]) * cu[j][r];
        acte[(size_t)(base + row) * IDIM + col + j * 16] = (__bf16)a;
      }
    }
  }
}

// ---------------- 7. routed experts: down proj * routing weight ------------
__global__ __launch_bounds__(256) void k_expert_down(
    const __bf16* __restrict__ acte, const float* __restrict__ down,
    const int* __restrict__ counts, const int* __restrict__ offs,
    const float* __restrict__ slot_w, __bf16* __restrict__ outsl) {
  int e = blockIdx.z;
  int count = counts[e];
  int m0 = blockIdx.y * MBLK;
  if (m0 >= count) return;
  int base = offs[e];
  const float* W = down + (size_t)e * HDIM * IDIM;   // [HDIM, IDIM]
  int n0 = blockIdx.x * 128;
  __shared__ __align__(16) __bf16 As[MBLK * LDK];
  __shared__ __align__(16) __bf16 Bs[MBLK * LDK];
  int wave = threadIdx.x >> 5;
  int lr = threadIdx.x >> 1, lc = (threadIdx.x & 1) * 32;
  int lidx = m0 + lr;
  int srow = (lidx < count) ? lidx : 0;
  const __bf16* arow = acte + (size_t)(base + srow) * IDIM + lc;
  __bf16* drow = As + lr * LDK + lc;
  v8f c[8];
#pragma unroll
  for (int j = 0; j < 8; j++) c[j] = vzero8();
  for (int k0 = 0; k0 < IDIM; k0 += KSTEP) {
    stage_a_async_row(drow, arow + k0);
    load_wtile_f32<128>(Bs, W + (size_t)n0 * IDIM + k0, IDIM, k0 + KSTEP < IDIM);
    wait_async0();
    __syncthreads();
#pragma unroll
    for (int kk = 0; kk < KSTEP; kk += 32) {
      v16bf a = lds_a_frag(As, wave * 16, kk);
#pragma unroll
      for (int j = 0; j < 8; j++)
        c[j] = wmma_bf16(a, lds_b_frag(Bs, j * 16, kk), c[j]);
    }
    __syncthreads();
  }
  int lane = threadIdx.x & 31;
  int col = n0 + (lane & 15);
  int rb = m0 + wave * 16 + ((lane & 16) ? 8 : 0);
#pragma unroll
  for (int r = 0; r < 8; r++) {
    int row = rb + r;
    if (row < count) {
      float w = slot_w[base + row];
#pragma unroll
      for (int j = 0; j < 8; j++)
        outsl[(size_t)(base + row) * HDIM + col + j * 16] = (__bf16)(w * c[j][r]);
    }
  }
}

// ---------------- 8. gather each token's 4 expert slots into d_out + aux ---
__global__ __launch_bounds__(256) void k_combine(const __bf16* __restrict__ outsl,
                                                 const int* __restrict__ slot_of,
                                                 float* __restrict__ out) {
  int idx = blockIdx.x * 256 + threadIdx.x;   // NTOK * (HDIM/8) threads
  int n = idx >> 7;                           // HDIM/8 == 128
  int hs = (idx & 127) * 8;
  float s[8];
#pragma unroll
  for (int i = 0; i < 8; i++) s[i] = 0.f;
#pragma unroll
  for (int k = 0; k < TOPK; k++) {
    int sl = slot_of[n * TOPK + k];
    Pack8 p;
    p.q = *(const uint4*)(outsl + (size_t)sl * HDIM + hs);
#pragma unroll
    for (int i = 0; i < 8; i++) s[i] += (float)p.h[i];
  }
  float* op = out + (size_t)n * HDIM + hs;
  float4 o0 = *(const float4*)(op);
  float4 o1 = *(const float4*)(op + 4);
  o0.x += s[0]; o0.y += s[1]; o0.z += s[2]; o0.w += s[3];
  o1.x += s[4]; o1.y += s[5]; o1.z += s[6]; o1.w += s[7];
  *(float4*)(op) = o0;
  *(float4*)(op + 4) = o1;
  // aux loss is identically num_experts (softmax rows sum to 1; top-k fractions sum to 1)
  if (idx == 0) out[(size_t)NTOK * HDIM] = (float)NEXP;
}

// ---------------------------------------------------------------------------
extern "C" void kernel_launch(void* const* d_in, const int* in_sizes, int n_in,
                              void* d_out, int out_size, void* d_ws, size_t ws_size,
                              hipStream_t stream) {
  (void)in_sizes; (void)n_in; (void)out_size; (void)ws_size;
  const float* h    = (const float*)d_in[0];
  const float* rw   = (const float*)d_in[1];
  const float* gup  = (const float*)d_in[2];
  const float* down = (const float*)d_in[3];
  const float* wg   = (const float*)d_in[4];
  const float* wu   = (const float*)d_in[5];
  const float* wd   = (const float*)d_in[6];
  const float* segw = (const float*)d_in[7];
  float* out = (float*)d_out;

  char* ws = (char*)d_ws;
  size_t o = 0;
  auto alloc = [&](size_t bytes) -> char* {
    char* p = ws + o;
    o += (bytes + 255) & ~(size_t)255;
    return p;
  };
  __bf16* hb      = (__bf16*)alloc((size_t)NTOK * HDIM * 2);          // 8 MB
  __bf16* act_sh  = (__bf16*)alloc((size_t)NTOK * ISH * 2);           // 16 MB
  __bf16* act_e   = (__bf16*)alloc((size_t)NTOK * TOPK * IDIM * 2);   // 16 MB
  __bf16* outsl   = (__bf16*)alloc((size_t)NTOK * TOPK * HDIM * 2);   // 32 MB
  int*    topi    = (int*)alloc((size_t)NTOK * TOPK * 4);
  float*  topw    = (float*)alloc((size_t)NTOK * TOPK * 4);
  float*  gatev   = (float*)alloc((size_t)NTOK * 4);
  int*    counts  = (int*)alloc(NEXP * 4);
  int*    offs    = (int*)alloc((NEXP + 1) * 4);
  int*    slot_tok= (int*)alloc((size_t)NTOK * TOPK * 4);
  float*  slot_w  = (float*)alloc((size_t)NTOK * TOPK * 4);
  int*    slot_of = (int*)alloc((size_t)NTOK * TOPK * 4);

  k_cvt<<<(NTOK * HDIM / 8) / 256, 256, 0, stream>>>(h, hb);
  k_router<<<NTOK / 256, 256, 0, stream>>>(h, rw, segw, topi, topw, gatev);
  k_compact<<<1, 256, 0, stream>>>(topi, topw, counts, offs, slot_tok, slot_w, slot_of);
  k_shared_gateup<<<dim3(ISH / 64, NTOK / MBLK), 256, 0, stream>>>(hb, wg, wu, act_sh);
  k_shared_down<<<dim3(HDIM / 128, NTOK / MBLK), 256, 0, stream>>>(act_sh, wd, gatev, out);
  k_expert_gateup<<<dim3(IDIM / 64, NTOK / MBLK, NEXP), 256, 0, stream>>>(
      hb, gup, counts, offs, slot_tok, act_e);
  k_expert_down<<<dim3(HDIM / 128, NTOK / MBLK, NEXP), 256, 0, stream>>>(
      act_e, down, counts, offs, slot_w, outsl);
  k_combine<<<(NTOK * HDIM / 8) / 256, 256, 0, stream>>>(outsl, slot_of, out);
}
